// Cross_attention_multi_71159018160210
// MI455X (gfx1250) — matrix operand order repaired and checked by reading, not compiled, not run on hardware
//
#include <hip/hip_runtime.h>
#include <hip/hip_bf16.h>

typedef __attribute__((ext_vector_type(16))) _Float16 v16h;
typedef __attribute__((ext_vector_type(8)))  float    v8f;

#define WMMA_F16(acc, af, bf) \
  acc = __builtin_amdgcn_wmma_f32_16x16x32_f16(false, af, false, bf, (short)0, acc, false, false)

// ---- geometry constants (from reference) ----
#define CCH   32          // channels
#define DD    36          // depth
#define HH    48
#define WW    48
#define HW    (HH*WW)     // 2304
#define NSPA  (DD*HW)     // 82944 spatial sites
#define PP    9
#define LROWS 1024        // (D/9)*(HW/9) = 4*256
#define KPAD  96          // 81 -> 96
#define NHID  162
#define NHPAD 176         // 162 -> 176 (11 n-tiles)
#define KHPAD 192         // 162 -> 192 (6 K-steps)
#define RTOT  (CCH*LROWS) // 32768 rows of t/a/b

// ---------------- fragment load helpers ----------------
// A fragment: row-major row pointer, K-chunk kk (multiple of 32).
// lane 0-15: K = kk+{0..7, 16..23}; lane 16-31: K = kk+{8..15, 24..31}
__device__ __forceinline__ v16h ldA(const _Float16* rowBase, int kk, int lo8) {
  union { v16h v; float4 q[2]; } u;
  u.q[0] = *(const float4*)(rowBase + kk + lo8);
  u.q[1] = *(const float4*)(rowBase + kk + 16 + lo8);
  return u.v;
}
// B fragment: column stored contiguously in K. lane 0-15: K=kk+{0..7, 16..23} ; lane16-31: K=kk+{8..15, 24..31}
__device__ __forceinline__ v16h ldB(const _Float16* colBase, int kk, int hi16) {
  union { v16h v; float4 q[2]; } u;
  u.q[0] = *(const float4*)(colBase + kk + (hi16 >> 1));
  u.q[1] = *(const float4*)(colBase + kk + 16 + (hi16 >> 1));
  return u.v;
}

// ---------------- kernel 1: weight repack ----------------
// wxa/wya: [27][32 co][32 ci] f16  (A-matrix rows, K=ci contiguous)
// w1b:     [176][96] f16  = zero-padded W1 (row n = hidden idx, k = patch idx)
// w2b:     [96][192] f16  = zero-padded W2 (row n = out idx,    k = hidden idx)
__global__ void k_prep(const float* __restrict__ Wx, const float* __restrict__ Wy,
                       const float* __restrict__ W1, const float* __restrict__ W2,
                       _Float16* __restrict__ wxa, _Float16* __restrict__ wya,
                       _Float16* __restrict__ w1b, _Float16* __restrict__ w2b) {
  const int tid = blockIdx.x * blockDim.x + threadIdx.x;
  const int stride = gridDim.x * blockDim.x;
  for (int i = tid; i < 27 * CCH * CCH; i += stride) {
    int t = i >> 10, co = (i >> 5) & 31, ci = i & 31;
    wxa[i] = (_Float16)Wx[(co * 32 + ci) * 27 + t];
    wya[i] = (_Float16)Wy[(co * 32 + ci) * 27 + t];
  }
  for (int i = tid; i < NHPAD * KPAD; i += stride) {
    int n = i / KPAD, k = i % KPAD;
    w1b[i] = (n < NHID && k < 81) ? (_Float16)W1[n * 81 + k] : (_Float16)0.0f;
  }
  for (int i = tid; i < KPAD * KHPAD; i += stride) {
    int n = i / KHPAD, k = i % KHPAD;
    w2b[i] = (n < 81 && k < NHID) ? (_Float16)W2[n * NHID + k] : (_Float16)0.0f;
  }
}

// ---------------- kernel 2: channel-minor f16 transpose ----------------
// src (C, D, H, W) f32 -> dst (s, C) f16 ; per-channel reads are fully coalesced
__global__ void k_transpose(const float* __restrict__ src, _Float16* __restrict__ dst) {
  int s = blockIdx.x * blockDim.x + threadIdx.x;
  if (s >= NSPA) return;
  union { _Float16 h[32]; float4 q[4]; } u;
#pragma unroll
  for (int c = 0; c < CCH; ++c) u.h[c] = (_Float16)src[(size_t)c * NSPA + s];
  float4* d = (float4*)(dst + (size_t)s * CCH);
  d[0] = u.q[0]; d[1] = u.q[1]; d[2] = u.q[2]; d[3] = u.q[3];
}

// ---------------- kernel 3: zero (padded unfold buffers) ----------------
__global__ void k_zero16(_Float16* p, int n128) { // n128 = number of 16B chunks
  int i = blockIdx.x * blockDim.x + threadIdx.x;
  int4 z = make_int4(0, 0, 0, 0);
  for (; i < n128; i += gridDim.x * blockDim.x) ((int4*)p)[i] = z;
}

// ---------------- kernel 4: conv3d as 27 tap-wise WMMA GEMMs ----------------
// One wave = one 16-wide w-tile (N) x all 32 out channels (2 M-tiles), K = 32 in-ch.
// Output scattered directly into unfolded (co*L + l, 96) f16 layout.
__global__ void k_conv(const _Float16* __restrict__ xt, const _Float16* __restrict__ wa,
                       const float* __restrict__ bias, _Float16* __restrict__ tout) {
  __shared__ _Float16 wsm[27 * CCH * CCH]; // 55296 B
  // cooperative stage of all 27 tap matrices into LDS (uint4 = 8 halfs)
  for (int i = threadIdx.x; i < (27 * CCH * CCH) / 8; i += blockDim.x)
    ((int4*)wsm)[i] = ((const int4*)wa)[i];
  __syncthreads();

  const int lane = threadIdx.x & 31;
  const int wave = threadIdx.x >> 5;
  const int tile = blockIdx.x * 8 + wave;      // grid sized exactly: 648*8 = 5184
  const int d    = tile / 144;
  const int rem  = tile % 144;
  const int h    = rem / 3;
  const int wbase = (rem % 3) * 16;

  const int n    = lane & 15;
  const int lo8  = (lane & 16) ? 8 : 0;
  const int hi16 = (lane & 16) ? 16 : 0;

  v8f acc0 = {}; v8f acc1 = {};
  const _Float16* arow0 = wsm + (lane & 15) * CCH;        // co = 0..15 row
  const _Float16* arow1 = wsm + (16 + (lane & 15)) * CCH; // co = 16..31 row

#pragma unroll
  for (int tap = 0; tap < 27; ++tap) {
    const int od = tap / 9 - 1, oh = (tap / 3) % 3 - 1, ow = tap % 3 - 1;
    const int sd = d + od, sh = h + oh, sw = wbase + n + ow;
    const bool valid = ((unsigned)sd < DD) & ((unsigned)sh < HH) & ((unsigned)sw < WW);
    v16h bf = {};
    if (valid) {
      const int sp = (sd * HH + sh) * WW + sw;
      bf = ldB(xt + (size_t)sp * CCH, 0, hi16);
    }
    v16h a0 = ldA(arow0 + tap * CCH * CCH, 0, lo8);
    v16h a1 = ldA(arow1 + tap * CCH * CCH, 0, lo8);
    WMMA_F16(acc0, a0, bf);
    WMMA_F16(acc1, a1, bf);
  }

  // bias add + scatter into unfolded layout: row = co*1024 + l, col = p
  const int w  = wbase + n;
  const int hw = h * WW + w;
  const int l  = (d / PP) * 256 + hw / PP;
  const int p  = (d % PP) * PP + hw % PP;
#pragma unroll
  for (int j = 0; j < 8; ++j) {
    const int co = j + ((lane & 16) ? 8 : 0);
    float v0 = acc0[j] + bias[co];
    float v1 = acc1[j] + bias[co + 16];
    tout[((size_t)(co)        * LROWS + l) * KPAD + p] = (_Float16)v0;
    tout[((size_t)(co + 16)   * LROWS + l) * KPAD + p] = (_Float16)v1;
  }
}

// ---------------- kernel 5: res_trans (81->162->81 MLP + LeakyReLU) ----------------
// One wave = 16 rows. GEMM1 -> wave-private LDS h tile (D->A transpose), GEMM2 -> act -> f16.
__global__ void k_restrans(const _Float16* __restrict__ t, const _Float16* __restrict__ w1b,
                           const _Float16* __restrict__ w2b, _Float16* __restrict__ aout) {
  __shared__ _Float16 hsm[8 * 16 * KHPAD]; // 8 waves * 16x192 f16 = 48 KB
  const int lane = threadIdx.x & 31;
  const int wave = threadIdx.x >> 5;
  const int rtile = blockIdx.x * 8 + wave;  // grid exactly 256*8 = 2048
  const int r0 = rtile * 16;
  const int lo8  = (lane & 16) ? 8 : 0;
  const int hi16 = (lane & 16) ? 16 : 0;
  const int hi8  = (lane & 16) ? 8 : 0;
  _Float16* hrow = hsm + wave * (16 * KHPAD);

  // zero this wave's h tile (covers pad cols 176..191)
  for (int i = lane; i < 16 * KHPAD; i += 32) hrow[i] = (_Float16)0.0f;

  // --- GEMM1: h = t @ W1^T ; M=16 rows, K=96, N=176 ---
  const _Float16* trow = t + (size_t)(r0 + (lane & 15)) * KPAD;
  v16h ta0 = ldA(trow, 0, lo8);
  v16h ta1 = ldA(trow, 32, lo8);
  v16h ta2 = ldA(trow, 64, lo8);
#pragma unroll
  for (int nt = 0; nt < 11; ++nt) {
    const _Float16* bcol = w1b + (size_t)(nt * 16 + (lane & 15)) * KPAD;
    v8f acc = {};
    WMMA_F16(acc, ta0, ldB(bcol, 0, hi16));
    WMMA_F16(acc, ta1, ldB(bcol, 32, hi16));
    WMMA_F16(acc, ta2, ldB(bcol, 64, hi16));
#pragma unroll
    for (int j = 0; j < 8; ++j) {
      const int M = j + hi8, N = lane & 15;
      hrow[M * KHPAD + nt * 16 + N] = (_Float16)acc[j];
    }
  }

  // --- GEMM2: o = leaky_relu(h @ W2^T) ; K=192, N=96 ---
  const _Float16* hArow = hrow + (lane & 15) * KHPAD;
  v16h ha[6];
#pragma unroll
  for (int kk = 0; kk < 6; ++kk) ha[kk] = ldA(hArow, kk * 32, lo8);
#pragma unroll
  for (int nt = 0; nt < 6; ++nt) {
    const _Float16* bcol = w2b + (size_t)(nt * 16 + (lane & 15)) * KHPAD;
    v8f acc = {};
#pragma unroll
    for (int kk = 0; kk < 6; ++kk) WMMA_F16(acc, ha[kk], ldB(bcol, kk * 32, hi16));
#pragma unroll
    for (int j = 0; j < 8; ++j) {
      const int M = j + hi8, N = lane & 15;
      float v = acc[j];
      v = v > 0.0f ? v : 0.2f * v;           // LeakyReLU(0.2)
      aout[((size_t)(r0 + M)) * KPAD + nt * 16 + N] = (_Float16)v;
    }
  }
}

// ---------------- kernel 6: attention a @ b^T / 81 ----------------
// One wave = one 16-row l-tile x 8 m-tiles of one channel; K = 96 (3 WMMA / tile).
__global__ void k_att(const _Float16* __restrict__ a, const _Float16* __restrict__ b,
                      float* __restrict__ out) {
  const int lane = threadIdx.x & 31;
  const int wave = threadIdx.x >> 5;
  const int wg = blockIdx.x * 8 + wave;  // grid exactly 2048*8 = 16384
  const int cl = wg >> 3;                // 0..2047 -> (c, ltile)
  const int mg = wg & 7;
  const int c = cl >> 6;
  const int ltile = cl & 63;
  const int lo8  = (lane & 16) ? 8 : 0;
  const int hi16 = (lane & 16) ? 16 : 0;
  const int hi8  = (lane & 16) ? 8 : 0;

  const _Float16* arow = a + ((size_t)(c * LROWS + ltile * 16 + (lane & 15))) * KPAD;
  v16h af0 = ldA(arow, 0, lo8);
  v16h af1 = ldA(arow, 32, lo8);
  v16h af2 = ldA(arow, 64, lo8);

  const float inv81 = 1.0f / 81.0f;
#pragma unroll
  for (int mt = 0; mt < 8; ++mt) {
    const int m0 = (mg * 8 + mt) * 16;
    const _Float16* bcol = b + ((size_t)(c * LROWS + m0 + (lane & 15))) * KPAD;
    if (mt < 7) __builtin_prefetch(bcol + 16 * KPAD, 0, 1);
    v8f acc = {};
    WMMA_F16(acc, af0, ldB(bcol, 0, hi16));
    WMMA_F16(acc, af1, ldB(bcol, 32, hi16));
    WMMA_F16(acc, af2, ldB(bcol, 64, hi16));
#pragma unroll
    for (int j = 0; j < 8; ++j) {
      const int M = j + hi8, N = lane & 15;
      out[((size_t)(c * LROWS) + ltile * 16 + M) * LROWS + m0 + N] = acc[j] * inv81;
    }
  }
}

// ---------------- host-side launch ----------------
extern "C" void kernel_launch(void* const* d_in, const int* in_sizes, int n_in,
                              void* d_out, int out_size, void* d_ws, size_t ws_size,
                              hipStream_t stream) {
  const float* x  = (const float*)d_in[0];
  const float* y  = (const float*)d_in[1];
  const float* Wx = (const float*)d_in[2];
  const float* bx = (const float*)d_in[3];
  const float* Wy = (const float*)d_in[4];
  const float* by = (const float*)d_in[5];
  const float* W1 = (const float*)d_in[6];
  const float* W2 = (const float*)d_in[7];
  float* out = (float*)d_out;

  // workspace layout (halfs), all offsets 16B aligned
  _Float16* ws = (_Float16*)d_ws;
  size_t o = 0;
  _Float16* xt  = ws + o; o += (size_t)NSPA * CCH;   // 2,654,208
  _Float16* yt  = ws + o; o += (size_t)NSPA * CCH;
  _Float16* tx  = ws + o; o += (size_t)RTOT * KPAD;  // 3,145,728
  _Float16* ty  = ws + o; o += (size_t)RTOT * KPAD;
  _Float16* ax  = ws + o; o += (size_t)RTOT * KPAD;
  _Float16* ay  = ws + o; o += (size_t)RTOT * KPAD;
  _Float16* wxa = ws + o; o += 27 * CCH * CCH;
  _Float16* wya = ws + o; o += 27 * CCH * CCH;
  _Float16* w1b = ws + o; o += NHPAD * KPAD;
  _Float16* w2b = ws + o; o += KPAD * KHPAD;

  k_prep<<<64, 256, 0, stream>>>(Wx, Wy, W1, W2, wxa, wya, w1b, w2b);
  k_transpose<<<(NSPA + 255) / 256, 256, 0, stream>>>(x, xt);
  k_transpose<<<(NSPA + 255) / 256, 256, 0, stream>>>(y, yt);
  // zero tx and ty (contiguous): 2 * RTOT * KPAD halfs / 8 halfs per 16B chunk
  k_zero16<<<1024, 256, 0, stream>>>(tx, (2 * RTOT * KPAD) / 8);
  k_conv<<<648, 256, 0, stream>>>(xt, wxa, bx, tx);
  k_conv<<<648, 256, 0, stream>>>(yt, wya, by, ty);
  k_restrans<<<256, 256, 0, stream>>>(tx, w1b, w2b, ax);
  k_restrans<<<256, 256, 0, stream>>>(ty, w1b, w2b, ay);
  k_att<<<2048, 256, 0, stream>>>(ax, ay, out);
}
